// QuadraticHeterogenousCrossAttention_26379689132068
// MI455X (gfx1250) — matrix operand order repaired and checked by reading, not compiled, not run on hardware
//
#include <hip/hip_runtime.h>
#include <hip/hip_bf16.h>
#include <cstdint>

typedef _Float16 half_t;
typedef __attribute__((ext_vector_type(16))) _Float16 v16h;
typedef __attribute__((ext_vector_type(8)))  float    v8f;
typedef __attribute__((ext_vector_type(2)))  _Float16 h2v;

constexpr int B_  = 4, T_ = 8, N_ = 128, D_ = 256, H_ = 8, DH_ = 32;
constexpr int S_  = T_ * N_;     // 1024
constexpr int BS_ = B_ * S_;     // 4096
constexpr int NF_ = 3;
constexpr float Y0C = 0.28209479177387814f;
constexpr float Y1C = 0.4886025119029199f;
// freqs_j = 1000^(-j/16) = exp2(-j * log2(1000)/16)
constexpr float L2K_O16 = 0.62286151779138045f;

// ---- WMMA operand lane maps (CDNA5 ISA 7.12.2, wave32) ----
// 16-bit A matrix 16x32: lane holds row M=lane%16; VGPR j packs K pair:
//   j<4 : K = 2j   + (lane>=16 ? 8 : 0)
//   j>=4: K = 16+2(j-4) + (lane>=16 ? 8 : 0)
__device__ __forceinline__ int a_k0(int lane, int j) {
  int hi = (lane >> 4) & 1;
  return (j < 4 ? 2 * j : 16 + 2 * (j - 4)) + hi * 8;
}
// 16-bit B matrix 32x16: lane holds col N=lane%16; VGPR j packs K pair:
//   j<4 : K = 2j   + (lane>=16 ? 8 : 0)
//   j>=4: K = 16+2(j-4) + (lane>=16 ? 8 : 0)
__device__ __forceinline__ int b_k0(int lane, int j) {
  return (j < 4 ? 2 * j : 16 + 2 * (j - 4)) + ((lane >> 4) & 1) * 8;
}
// C/D f32 16x16: VGPR r -> (M = r + (lane>=16?8:0), N = lane%16)

__device__ __forceinline__ v8f wmma_f16(v16h a, v16h b, v8f c) {
  return __builtin_amdgcn_wmma_f32_16x16x32_f16(false, a, false, b, (short)0, c,
                                                false, false);
}

// =====================  Q projection + RoPE  =====================
// q = q_data(BS,D) @ Wq + bq ; RoPE per head ; store f16 row-major
__global__ __launch_bounds__(128) void qproj_kernel(
    const float* __restrict__ qdata, const float* __restrict__ Wq,
    const float* __restrict__ bq, const unsigned char* __restrict__ mask,
    half_t* __restrict__ Qws) {
  const int lane = threadIdx.x & 31;
  const int wv   = threadIdx.x >> 5;
  const int mtile = blockIdx.x;
  const int ntile = blockIdx.y * 4 + wv;
  const int m  = lane & 15;
  const int nl = lane & 15;
  const int hi = (lane >> 4) & 1;

  v8f c = {};
#pragma unroll
  for (int kc = 0; kc < D_ / 32; ++kc) {
    const int kb = kc * 32;
    v16h a = {}, b = {};
#pragma unroll
    for (int j = 0; j < 8; ++j) {
      const int ka = a_k0(lane, j);
      const float2 av =
          *(const float2*)(qdata + (size_t)(mtile * 16 + m) * D_ + kb + ka);
      a[2 * j] = (half_t)av.x; a[2 * j + 1] = (half_t)av.y;
      const int kbb = b_k0(lane, j);
      b[2 * j]     = (half_t)Wq[(size_t)(kb + kbb) * D_ + ntile * 16 + nl];
      b[2 * j + 1] = (half_t)Wq[(size_t)(kb + kbb + 1) * D_ + ntile * 16 + nl];
    }
    c = wmma_f16(a, b, c);
  }
  const int col   = ntile * 16 + nl;
  const float bqv = bq[col];
  const int jr    = (col & 31) >> 1;
  const float frq = exp2f(-(float)jr * L2K_O16);
  const bool evn  = (col & 1) == 0;
#pragma unroll
  for (int r = 0; r < 8; ++r) {
    const int row = mtile * 16 + r + hi * 8;
    const int bb = row / S_, s = row % S_;
    const int t = s / N_, nn = s % N_;
    const bool mv  = mask[bb * N_ + nn] != 0;
    const float ang = (float)t * frq;
    const float cs = mv ? cosf(ang) : 1.0f;
    const float sn = mv ? sinf(ang) : 0.0f;
    const float v  = c[r] + bqv;
    const float pv = __shfl_xor(v, 1, 32);  // partner column of RoPE pair
    const float ov = v * cs + (evn ? -pv * sn : pv * sn);
    Qws[(size_t)row * D_ + col] = (half_t)ov;
  }
}

// =====================  KV projection (3 features)  =====================
// kv = feat(BS,D) @ Wkv[f](D,2D) + bkv[f]; K half gets RoPE -> Kws row-major;
// V half stored transposed: Vt[f][b][d][s]  (PV B-operand contiguity)
__global__ __launch_bounds__(128) void kvproj_kernel(
    const float* __restrict__ x0, const float* __restrict__ v0,
    const float* __restrict__ cfeat, const float* __restrict__ Wkv,
    const float* __restrict__ bkv, const unsigned char* __restrict__ mask,
    half_t* __restrict__ Kws, half_t* __restrict__ Vtws) {
  const int f = blockIdx.z;
  const float* feat = (f == 0) ? x0 : (f == 1) ? v0 : cfeat;
  const float* Wf = Wkv + (size_t)f * D_ * 2 * D_;
  const float* bf = bkv + (size_t)f * 2 * D_;

  const int lane = threadIdx.x & 31;
  const int wv   = threadIdx.x >> 5;
  const int mtile = blockIdx.x;
  const int ntile = blockIdx.y * 4 + wv;  // 0..31 over 512 cols
  const int m  = lane & 15;
  const int nl = lane & 15;
  const int hi = (lane >> 4) & 1;

  v8f c = {};
#pragma unroll
  for (int kc = 0; kc < D_ / 32; ++kc) {
    const int kb = kc * 32;
    v16h a = {}, b = {};
#pragma unroll
    for (int j = 0; j < 8; ++j) {
      const int ka = a_k0(lane, j);
      const float2 av =
          *(const float2*)(feat + (size_t)(mtile * 16 + m) * D_ + kb + ka);
      a[2 * j] = (half_t)av.x; a[2 * j + 1] = (half_t)av.y;
      const int kbb = b_k0(lane, j);
      b[2 * j]     = (half_t)Wf[(size_t)(kb + kbb) * (2 * D_) + ntile * 16 + nl];
      b[2 * j + 1] = (half_t)Wf[(size_t)(kb + kbb + 1) * (2 * D_) + ntile * 16 + nl];
    }
    c = wmma_f16(a, b, c);
  }
  const int col = ntile * 16 + nl;  // 0..511
  const float bv = bf[col];
  if (col < D_) {  // K half: RoPE (branch uniform per wave)
    const int jr    = (col & 31) >> 1;
    const float frq = exp2f(-(float)jr * L2K_O16);
    const bool evn  = (col & 1) == 0;
#pragma unroll
    for (int r = 0; r < 8; ++r) {
      const int row = mtile * 16 + r + hi * 8;
      const int bb = row / S_, s = row % S_;
      const int t = s / N_, nn = s % N_;
      const bool mv  = mask[bb * N_ + nn] != 0;
      const float ang = (float)t * frq;
      const float cs = mv ? cosf(ang) : 1.0f;
      const float sn = mv ? sinf(ang) : 0.0f;
      const float v  = c[r] + bv;
      const float pv = __shfl_xor(v, 1, 32);
      const float ov = v * cs + (evn ? -pv * sn : pv * sn);
      Kws[((size_t)f * BS_ + row) * D_ + col] = (half_t)ov;
    }
  } else {  // V half: store transposed
    const int d = col - D_;
#pragma unroll
    for (int r = 0; r < 8; ++r) {
      const int row = mtile * 16 + r + hi * 8;
      const int bb = row / S_, s = row % S_;
      Vtws[(((size_t)f * B_ + bb) * D_ + d) * S_ + s] = (half_t)(c[r] + bv);
    }
  }
}

// =====================  Fused flash attention + SH bias  ==================
// Block = 256 threads (8 waves); wave w -> head w; block -> (b, 16-query tile)
__global__ __launch_bounds__(256) void attn_kernel(
    const float* __restrict__ x0, const unsigned char* __restrict__ mask,
    const float* __restrict__ fw, const float* __restrict__ denom,
    const float* __restrict__ W1, const float* __restrict__ b1,
    const float* __restrict__ W2, const float* __restrict__ b2,
    const float* __restrict__ W3, const float* __restrict__ b3,
    const half_t* __restrict__ Qws, const half_t* __restrict__ Kws,
    const half_t* __restrict__ Vtws, half_t* __restrict__ Ows) {
  __shared__ float sW1[64], sB1[16], sW2[256], sB2[16], sW3[128], sB3[8];
  __shared__ float qc[16][3];
  __shared__ float kco[32][3];
  __shared__ float h2buf[16][32][16];   // shared MLP hidden, all heads reuse
  __shared__ half_t pbuf[8][16][32];    // per-wave P C-layout -> A-layout bounce

  const int tid  = threadIdx.x;
  const int lane = tid & 31;
  const int h    = tid >> 5;           // head == wave
  const int b    = blockIdx.x >> 6;
  const int qbase = (blockIdx.x & 63) * 16;
  const int nl = lane & 15;
  const int hi = (lane >> 4) & 1;

  for (int i = tid; i < 64;  i += 256) sW1[i] = W1[i];
  for (int i = tid; i < 256; i += 256) sW2[i] = W2[i];
  for (int i = tid; i < 128; i += 256) sW3[i] = W3[i];
  if (tid < 16) { sB1[tid] = b1[tid]; sB2[tid] = b2[tid]; }
  if (tid < 8)  sB3[tid] = b3[tid];
  if (tid < 16) {
    const float* cp = x0 + ((size_t)b * S_ + qbase + tid) * D_;
    qc[tid][0] = cp[0]; qc[tid][1] = cp[1]; qc[tid][2] = cp[2];
  }

  // softmax(feature_weights)
  const float g0 = fw[0], g1 = fw[1], g2 = fw[2];
  const float gm = fmaxf(g0, fmaxf(g1, g2));
  const float e0 = __expf(g0 - gm), e1 = __expf(g1 - gm), e2 = __expf(g2 - gm);
  const float gs = 1.0f / (e0 + e1 + e2);
  const float gate[3] = {e0 * gs, e1 * gs, e2 * gs};
  const float inv_den = 1.0f / denom[h];

  // Q tile in A layout (once)
  v16h qa = {};
#pragma unroll
  for (int j = 0; j < 8; ++j) {
    const int k0 = a_k0(lane, j);
    const h2v p =
        *(const h2v*)(Qws + ((size_t)b * S_ + qbase + nl) * D_ + h * DH_ + k0);
    qa[2 * j] = p.x; qa[2 * j + 1] = p.y;
  }

  const v8f zf = {0.f, 0.f, 0.f, 0.f, 0.f, 0.f, 0.f, 0.f};
  v8f acc0[NF_], acc1[NF_];
  float mrun[NF_][8], lrun[NF_][8];
#pragma unroll
  for (int f = 0; f < NF_; ++f) {
    acc0[f] = zf; acc1[f] = zf;
#pragma unroll
    for (int r = 0; r < 8; ++r) { mrun[f][r] = -1e30f; lrun[f][r] = 0.f; }
  }

  for (int kt = 0; kt < S_ / 32; ++kt) {
    const int kbase = kt * 32;
    __syncthreads();  // previous-iteration h2buf readers done
    if (tid < 32) {
      const float* cp = x0 + ((size_t)b * S_ + kbase + tid) * D_;
      kco[tid][0] = cp[0]; kco[tid][1] = cp[1]; kco[tid][2] = cp[2];
    }
    __syncthreads();
    // SH MLP hidden for 16x32 (q,k) pairs, shared across heads & features
    for (int pp = tid; pp < 512; pp += 256) {
      const int qi = pp >> 5, ki = pp & 31;
      const float rx = qc[qi][0] - kco[ki][0];
      const float ry = qc[qi][1] - kco[ki][1];
      const float rz = qc[qi][2] - kco[ki][2];
      const float nrm = sqrtf(rx * rx + ry * ry + rz * rz);
      const float inv = 1.0f / (nrm + 1e-6f);
      const float sh1 = Y1C * ry * inv, sh2 = Y1C * rz * inv, sh3 = Y1C * rx * inv;
      float h1v[16];
#pragma unroll
      for (int j = 0; j < 16; ++j) {
        const float a = sB1[j] + Y0C * sW1[j] + sh1 * sW1[16 + j] +
                        sh2 * sW1[32 + j] + sh3 * sW1[48 + j];
        h1v[j] = a / (1.0f + __expf(-a));  // silu
      }
#pragma unroll
      for (int j = 0; j < 16; ++j) {
        float a = sB2[j];
#pragma unroll
        for (int i2 = 0; i2 < 16; ++i2) a += h1v[i2] * sW2[i2 * 16 + j];
        h2buf[qi][ki][j] = a / (1.0f + __expf(-a));
      }
    }
    __syncthreads();

    const bool mk0 = mask[b * N_ + ((kbase + nl) & (N_ - 1))] != 0;
    const bool mk1 = mask[b * N_ + ((kbase + 16 + nl) & (N_ - 1))] != 0;

    // per-head bias tiles from shared hidden (C layout)
    float bias0[8], bias1[8];
#pragma unroll
    for (int r = 0; r < 8; ++r) {
      const int M = r + hi * 8;
      float a0 = sB3[h], a1 = sB3[h];
#pragma unroll
      for (int j = 0; j < 16; ++j) {
        a0 += h2buf[M][nl][j] * sW3[j * H_ + h];
        a1 += h2buf[M][nl + 16][j] * sW3[j * H_ + h];
      }
      bias0[r] = a0; bias1[r] = a1;
    }

#pragma unroll
    for (int f = 0; f < NF_; ++f) {
      // K B-tiles (dh x 16 keys) x2
      const half_t* Kf = Kws + ((size_t)f * B_ + b) * (size_t)S_ * D_;
      v16h kb0 = {}, kb1 = {};
#pragma unroll
      for (int j = 0; j < 8; ++j) {
        const int kd = b_k0(lane, j);
        const h2v p0 = *(const h2v*)(Kf + (size_t)(kbase + nl) * D_ + h * DH_ + kd);
        kb0[2 * j] = p0.x; kb0[2 * j + 1] = p0.y;
        const h2v p1 =
            *(const h2v*)(Kf + (size_t)(kbase + 16 + nl) * D_ + h * DH_ + kd);
        kb1[2 * j] = p1.x; kb1[2 * j + 1] = p1.y;
      }
      v8f s0 = wmma_f16(qa, kb0, zf);
      v8f s1 = wmma_f16(qa, kb1, zf);

      // online softmax (rows split across lane halves)
      float p0v[8], p1v[8];
#pragma unroll
      for (int r = 0; r < 8; ++r) {
        const float sc0 = mk0 ? (s0[r] * inv_den + bias0[r]) : -1e9f;
        const float sc1 = mk1 ? (s1[r] * inv_den + bias1[r]) : -1e9f;
        float rm = fmaxf(sc0, sc1);
        rm = fmaxf(rm, __shfl_xor(rm, 1, 32));
        rm = fmaxf(rm, __shfl_xor(rm, 2, 32));
        rm = fmaxf(rm, __shfl_xor(rm, 4, 32));
        rm = fmaxf(rm, __shfl_xor(rm, 8, 32));
        const float mn = fmaxf(mrun[f][r], rm);
        const float sc = __expf(mrun[f][r] - mn);
        mrun[f][r] = mn;
        const float q0 = __expf(sc0 - mn), q1 = __expf(sc1 - mn);
        float rs = q0 + q1;
        rs += __shfl_xor(rs, 1, 32);
        rs += __shfl_xor(rs, 2, 32);
        rs += __shfl_xor(rs, 4, 32);
        rs += __shfl_xor(rs, 8, 32);
        lrun[f][r] = lrun[f][r] * sc + rs;
        acc0[f][r] *= sc; acc1[f][r] *= sc;
        p0v[r] = q0; p1v[r] = q1;
      }
      // P: C layout -> LDS -> A layout
      asm volatile("s_wait_dscnt 0" ::: "memory");
#pragma unroll
      for (int r = 0; r < 8; ++r) {
        const int M = r + hi * 8;
        pbuf[h][M][nl]      = (half_t)p0v[r];
        pbuf[h][M][16 + nl] = (half_t)p1v[r];
      }
      asm volatile("s_wait_dscnt 0" ::: "memory");
      v16h pa = {};
#pragma unroll
      for (int j = 0; j < 8; ++j) {
        const int k0 = a_k0(lane, j);
        const h2v p = *(const h2v*)&pbuf[h][nl][k0];
        pa[2 * j] = p.x; pa[2 * j + 1] = p.y;
      }
      // V B-tiles (32 keys x 16 dh) x2, from transposed V
      const half_t* Vf = Vtws + ((size_t)f * B_ + b) * (size_t)D_ * S_;
      v16h vb0 = {}, vb1 = {};
#pragma unroll
      for (int j = 0; j < 8; ++j) {
        const int kk = b_k0(lane, j);
        const h2v p0 = *(const h2v*)(Vf + (size_t)(h * DH_ + nl) * S_ + kbase + kk);
        vb0[2 * j] = p0.x; vb0[2 * j + 1] = p0.y;
        const h2v p1 =
            *(const h2v*)(Vf + (size_t)(h * DH_ + 16 + nl) * S_ + kbase + kk);
        vb1[2 * j] = p1.x; vb1[2 * j + 1] = p1.y;
      }
      acc0[f] = wmma_f16(pa, vb0, acc0[f]);
      acc1[f] = wmma_f16(pa, vb1, acc1[f]);
    }
  }

  // gated combine + store attention output (f16 for final GEMM)
#pragma unroll
  for (int r = 0; r < 8; ++r) {
    float o0 = 0.f, o1 = 0.f;
#pragma unroll
    for (int f = 0; f < NF_; ++f) {
      const float il = 1.0f / lrun[f][r];
      o0 += gate[f] * acc0[f][r] * il;
      o1 += gate[f] * acc1[f][r] * il;
    }
    const size_t row = (size_t)b * S_ + qbase + r + hi * 8;
    Ows[row * D_ + h * DH_ + nl]      = (half_t)o0;
    Ows[row * D_ + h * DH_ + 16 + nl] = (half_t)o1;
  }
}

// =====================  Output projection  =====================
__global__ __launch_bounds__(128) void oproj_kernel(
    const half_t* __restrict__ Ows, const float* __restrict__ Wo,
    const float* __restrict__ bo, float* __restrict__ out) {
  const int lane = threadIdx.x & 31;
  const int wv   = threadIdx.x >> 5;
  const int mtile = blockIdx.x;
  const int ntile = blockIdx.y * 4 + wv;
  const int m  = lane & 15;
  const int nl = lane & 15;
  const int hi = (lane >> 4) & 1;

  v8f c = {};
#pragma unroll
  for (int kc = 0; kc < D_ / 32; ++kc) {
    const int kb = kc * 32;
    v16h a = {}, b = {};
#pragma unroll
    for (int j = 0; j < 8; ++j) {
      const int ka = a_k0(lane, j);
      const h2v av = *(const h2v*)(Ows + (size_t)(mtile * 16 + m) * D_ + kb + ka);
      a[2 * j] = av.x; a[2 * j + 1] = av.y;
      const int kbb = b_k0(lane, j);
      b[2 * j]     = (half_t)Wo[(size_t)(kb + kbb) * D_ + ntile * 16 + nl];
      b[2 * j + 1] = (half_t)Wo[(size_t)(kb + kbb + 1) * D_ + ntile * 16 + nl];
    }
    c = wmma_f16(a, b, c);
  }
  const int col = ntile * 16 + nl;
  const float bv = bo[col];
#pragma unroll
  for (int r = 0; r < 8; ++r) {
    const int row = mtile * 16 + r + hi * 8;
    out[(size_t)row * D_ + col] = c[r] + bv;
  }
}

extern "C" void kernel_launch(void* const* d_in, const int* in_sizes, int n_in,
                              void* d_out, int out_size, void* d_ws,
                              size_t ws_size, hipStream_t stream) {
  (void)in_sizes; (void)n_in; (void)out_size; (void)ws_size;
  const float* x0   = (const float*)d_in[0];
  const float* v0   = (const float*)d_in[1];
  const float* cf   = (const float*)d_in[2];
  const float* qd   = (const float*)d_in[3];
  const unsigned char* mask = (const unsigned char*)d_in[4];
  const float* Wq   = (const float*)d_in[5];
  const float* bq   = (const float*)d_in[6];
  const float* Wkv  = (const float*)d_in[7];
  const float* bkv  = (const float*)d_in[8];
  const float* Wo   = (const float*)d_in[9];
  const float* bo   = (const float*)d_in[10];
  const float* fw   = (const float*)d_in[11];
  const float* den  = (const float*)d_in[12];
  const float* W1   = (const float*)d_in[13];
  const float* b1   = (const float*)d_in[14];
  const float* W2   = (const float*)d_in[15];
  const float* b2   = (const float*)d_in[16];
  const float* W3   = (const float*)d_in[17];
  const float* b3   = (const float*)d_in[18];

  half_t* Qws  = (half_t*)d_ws;                       // 1M halves
  half_t* Kws  = Qws + (size_t)BS_ * D_;              // 3M halves
  half_t* Vtws = Kws + (size_t)NF_ * BS_ * D_;        // 3M halves
  half_t* Ows  = Vtws + (size_t)NF_ * BS_ * D_;       // 1M halves
  float* out = (float*)d_out;

  qproj_kernel<<<dim3(BS_ / 16, D_ / 64), 128, 0, stream>>>(qd, Wq, bq, mask, Qws);
  kvproj_kernel<<<dim3(BS_ / 16, (2 * D_) / 64, NF_), 128, 0, stream>>>(
      x0, v0, cf, Wkv, bkv, mask, Kws, Vtws);
  attn_kernel<<<dim3(B_ * (S_ / 16)), 256, 0, stream>>>(
      x0, mask, fw, den, W1, b1, W2, b2, W3, b3, Qws, Kws, Vtws, Ows);
  oproj_kernel<<<dim3(BS_ / 16, D_ / 64), 128, 0, stream>>>(Ows, Wo, bo, out);
}
